// GCNEncoder_55645596287657
// MI455X (gfx1250) — matrix operand order repaired and checked by reading, not compiled, not run on hardware
//
#include <hip/hip_runtime.h>
#include <hip/hip_bf16.h>
#include <math.h>

// ---------------------------------------------------------------------------
// Problem constants (from reference)
// ---------------------------------------------------------------------------
#define Bn    32
#define Cc    64      // nodes per graph
#define FIN   200
#define KP    224     // FIN padded to multiple of 32
#define TPn   64
#define Gg    (Bn * TPn)   // 2048 graphs
#define Ee    512
#define NH    256
#define NO    128
#define KSEL  58      // ceil(0.9*64)
#define EPSC  1e-5f
#define ZOFF  (Gg * KSEL * NO)   // 15,204,352 floats

typedef __attribute__((ext_vector_type(16))) _Float16 v16h;
typedef __attribute__((ext_vector_type(8)))  float    v8f;

// ---------------------------------------------------------------------------
// WMMA helpers (CDNA5, wave32). D = A(16x32 f16) * B(32x16 f16) + C(16x16 f32)
// A layout (ISA 7.12.2): lane L holds row M=L&15; halves [0..7]=K(koff..koff+7),
// [8..15]=K(koff+16..koff+23), koff = 8*(L>=16).
// B layout (N-major storage [N][K]): lane L holds col N=L&15; halves
// [0..7]=K(koff..koff+7), [8..15]=K(koff+16..koff+23), koff = kb + 8*(L>=16).
// C/D: c[v] = D[M = v + 8*(L>=16)][N = L&15].
// ---------------------------------------------------------------------------
__device__ inline v8f wmma_f16(v16h a, v16h b, v8f c) {
  return __builtin_amdgcn_wmma_f32_16x16x32_f16(
      /*neg_a=*/false, a, /*neg_b=*/false, b,
      /*c_mod=*/(short)0, c, /*reuse_a=*/false, /*reuse_b=*/false);
}

template <typename P>
__device__ inline v16h load_frag_A(P base, int ld, int mbase, int kb, int lane) {
  int m    = mbase + (lane & 15);
  int koff = kb + ((lane >> 4) << 3);
  P p = base + m * ld + koff;
  v16h f;
#pragma unroll
  for (int i = 0; i < 8; ++i) { f[i] = p[i]; f[i + 8] = p[i + 16]; }
  return f;
}

template <typename P>
__device__ inline v16h load_frag_B(P base, int ld, int nbase, int kb, int lane) {
  int n    = nbase + (lane & 15);
  int koff = kb + ((lane >> 4) << 3);
  P p = base + n * ld + koff;
  v16h f;
#pragma unroll
  for (int i = 0; i < 8; ++i) { f[i] = p[i]; f[i + 8] = p[i + 16]; }
  return f;
}

// ---------------------------------------------------------------------------
// Prep kernel (1 block, 256 threads): build shared dense aggregation matrices
// M1 (weighted edges) and Mp (unit edges), and f16 N-major copies of W1/W2.
// ---------------------------------------------------------------------------
__global__ void gcn_prep_kernel(const int* __restrict__ ei,
                                const float* __restrict__ ew,
                                const float* __restrict__ W1,
                                const float* __restrict__ W2,
                                float* __restrict__ M1f,
                                float* __restrict__ Mpf,
                                _Float16* __restrict__ W1h,
                                _Float16* __restrict__ W2h) {
  __shared__ float deg1[Cc], degp[Cc], dinv1[Cc], dinvp[Cc];
  __shared__ float M1L[Cc * Cc], MpL[Cc * Cc];
  const int tid = threadIdx.x;

  for (int i = tid; i < Cc * Cc; i += 256) { M1L[i] = 0.f; MpL[i] = 0.f; }
  if (tid < Cc) { deg1[tid] = 0.f; degp[tid] = 0.f; }
  __syncthreads();

  for (int e = tid; e < Ee; e += 256) {
    int c = ei[Ee + e];
    atomicAdd(&deg1[c], ew[e]);
    atomicAdd(&degp[c], 1.0f);
  }
  __syncthreads();
  if (tid < Cc) {
    dinv1[tid] = rsqrtf(deg1[tid] + 1.0f);
    dinvp[tid] = rsqrtf(degp[tid] + 1.0f);
  }
  __syncthreads();
  for (int e = tid; e < Ee; e += 256) {
    int r = ei[e], c = ei[Ee + e];
    atomicAdd(&M1L[c * Cc + r], dinv1[r] * ew[e] * dinv1[c]);
    atomicAdd(&MpL[c * Cc + r], dinvp[r] * dinvp[c]);
  }
  __syncthreads();
  if (tid < Cc) {
    M1L[tid * Cc + tid] += dinv1[tid] * dinv1[tid];
    MpL[tid * Cc + tid] += dinvp[tid] * dinvp[tid];
  }
  __syncthreads();
  for (int i = tid; i < Cc * Cc; i += 256) { M1f[i] = M1L[i]; Mpf[i] = MpL[i]; }

  // W1 (FIN x NH) -> W1h [NH][KP] f16, K padded with zeros
  for (int i = tid; i < NH * KP; i += 256) {
    int n = i / KP, f = i % KP;
    W1h[i] = (f < FIN) ? (_Float16)W1[f * NH + n] : (_Float16)0.0f;
  }
  // W2 (NH x NO) -> W2h [NO][NH] f16
  for (int i = tid; i < NO * NH; i += 256) {
    int n = i >> 8, k = i & 255;
    W2h[i] = (_Float16)W2[k * NO + n];
  }
}

// ---------------------------------------------------------------------------
// Main kernel: one workgroup (128 threads = 4 wave32) per graph.
// Dynamic LDS layout (92160 B total):
//   R0 @ 0      (32768): x_gh [64][224] f16  -> x1h  [64][256] f16
//   R1 @ 32768  (32768): xw1hT[256][64] f16  -> xph  [64][256] f16
//   R2 @ 65536  (16384): M2f  [64][64]  f32  -> xw2hT[128][64] f16
//   R3 @ 81920  ( 8192): M1h  [64][64]  f16  -> M2h  [64][64]  f16
//   small @ 90112: tvec[64] svec[64] svals[64] deg2[64] zsum[128] sidx[64] pos[64]
// ---------------------------------------------------------------------------
#define SMEM_BYTES 92160

__global__ void gcn_main_kernel(const float* __restrict__ h,
                                const int* __restrict__ ei,
                                const float* __restrict__ ew,
                                const float* __restrict__ b1,
                                const float* __restrict__ g1,
                                const float* __restrict__ be1,
                                const float* __restrict__ rm1,
                                const float* __restrict__ rv1,
                                const float* __restrict__ a1,
                                const float* __restrict__ Wp,
                                const float* __restrict__ bp,
                                const float* __restrict__ b2,
                                const float* __restrict__ g2,
                                const float* __restrict__ be2,
                                const float* __restrict__ rm2,
                                const float* __restrict__ rv2,
                                const float* __restrict__ a2,
                                const float* __restrict__ M1f,
                                const float* __restrict__ Mpf,
                                const _Float16* __restrict__ W1h,
                                const _Float16* __restrict__ W2h,
                                float* __restrict__ out) {
  extern __shared__ char smem[];
  _Float16* R0    = (_Float16*)(smem);            // x_gh then x1h
  _Float16* R1    = (_Float16*)(smem + 32768);    // xw1hT then xph
  float*    M2f   = (float*)   (smem + 65536);
  _Float16* R2h   = (_Float16*)(smem + 65536);    // xw2hT (after M2f retired)
  _Float16* R3    = (_Float16*)(smem + 81920);    // M1h then M2h
  float*    tvec  = (float*)(smem + 90112);
  float*    svec  = (float*)(smem + 90368);
  float*    svals = (float*)(smem + 90624);
  float*    deg2  = (float*)(smem + 90880);
  float*    zsum  = (float*)(smem + 91136);       // 128 floats
  int*      sidx  = (int*)  (smem + 91648);
  int*      pos   = (int*)  (smem + 91904);

  const int g    = blockIdx.x;
  const int tid  = threadIdx.x;
  const int lane = tid & 31;
  const int wv   = tid >> 5;       // wave id 0..3 -> M-tile
  const int bb   = g >> 6;         // batch index
  const int tt   = g & 63;         // time index

  // ---- stage x_g (h[b, c, f, t]) as f16 [64][224], load M1h, zero zsum ----
  zsum[tid] = 0.f;
  for (int i = tid; i < Cc * KP; i += 128) {
    int c = i / KP, f = i % KP;
    R0[i] = (f < FIN)
              ? (_Float16)h[(size_t)bb * (Cc * FIN * TPn) + c * (FIN * TPn) + f * TPn + tt]
              : (_Float16)0.0f;
  }
  for (int i = tid; i < Cc * Cc; i += 128) R3[i] = (_Float16)M1f[i];
  __syncthreads();

  // ---- GEMM1: xw1 = x_g(64x224) @ W1h^T  -> xw1hT[feat][node] (f16) ----
  for (int nt = 0; nt < 16; ++nt) {
    v8f acc = {};
#pragma unroll
    for (int kb = 0; kb < 7; ++kb) {
      v16h a = load_frag_A((const _Float16*)R0, KP, 16 * wv, 32 * kb, lane);
      v16h b = load_frag_B(W1h, KP, 16 * nt, 32 * kb, lane);
      acc = wmma_f16(a, b, acc);
    }
#pragma unroll
    for (int v = 0; v < 8; ++v) {
      int node = 16 * wv + v + ((lane >> 4) << 3);
      int f    = 16 * nt + (lane & 15);
      R1[f * Cc + node] = (_Float16)acc[v];
    }
  }
  __syncthreads();

  // ---- Agg1: x1 = M1 @ xw1, + b1, BN1, PReLU -> x1h row-major [64][256] ----
  const float alpha1 = a1[0];
  for (int nt = 0; nt < 16; ++nt) {
    v8f acc = {};
#pragma unroll
    for (int kb = 0; kb < 2; ++kb) {
      v16h a = load_frag_A((const _Float16*)R3, Cc, 16 * wv, 32 * kb, lane);
      v16h b = load_frag_B((const _Float16*)R1, Cc, 16 * nt, 32 * kb, lane);
      acc = wmma_f16(a, b, acc);
    }
    int f = 16 * nt + (lane & 15);
    float sc = g1[f] * rsqrtf(rv1[f] + EPSC);
    float sh = be1[f] - rm1[f] * sc;
    float bv = b1[f];
#pragma unroll
    for (int v = 0; v < 8; ++v) {
      int node = 16 * wv + v + ((lane >> 4) << 3);
      float x = acc[v] + bv;
      x = x * sc + sh;
      x = (x >= 0.f) ? x : alpha1 * x;
      R0[node * NH + f] = (_Float16)x;
    }
  }
  __syncthreads();

  // ---- Pooling score: s = tanh(Mp @ (x1 @ Wp) + bp) ----
  if (tid < Cc) {
    float t = 0.f;
    for (int k = 0; k < NH; ++k) t += (float)R0[tid * NH + k] * Wp[k];
    tvec[tid] = t;
  }
  __syncthreads();
  if (tid < Cc) {
    float sc = bp[0];
    for (int r = 0; r < Cc; ++r) sc += Mpf[tid * Cc + r] * tvec[r];
    svec[tid] = tanhf(sc);
  }
  __syncthreads();

  // ---- top-k (k=58), stable descending (ties -> lower index first) ----
  if (tid < Cc) {
    float sc = svec[tid];
    int rank = 0;
    for (int j = 0; j < Cc; ++j) {
      float sj = svec[j];
      rank += (sj > sc) || (sj == sc && j < tid);
    }
    if (rank < KSEL) { pos[tid] = rank; sidx[rank] = tid; svals[rank] = sc; }
    else             { pos[tid] = -1; }
  }
  __syncthreads();

  // ---- gather/scale xp = x1[idx]*vals -> xph [64][256]; zero M2f/deg2 ----
  for (int i = tid; i < Cc * NH; i += 128) {
    int j = i >> 8, f = i & 255;
    R1[i] = (j < KSEL) ? (_Float16)((float)R0[sidx[j] * NH + f] * svals[j])
                       : (_Float16)0.0f;
  }
  for (int i = tid; i < Cc * Cc; i += 128) M2f[i] = 0.f;
  if (tid < Cc) deg2[tid] = 0.f;
  __syncthreads();

  // ---- per-graph M2 from remapped edges ----
  for (int e = tid; e < Ee; e += 128) {
    int nr = pos[ei[e]], nc = pos[ei[Ee + e]];
    if (nr >= 0 && nc >= 0) atomicAdd(&deg2[nc], ew[e]);
  }
  __syncthreads();
  if (tid < Cc) deg2[tid] = rsqrtf(deg2[tid] + 1.0f);   // now holds dinv2
  __syncthreads();
  for (int e = tid; e < Ee; e += 128) {
    int nr = pos[ei[e]], nc = pos[ei[Ee + e]];
    if (nr >= 0 && nc >= 0)
      atomicAdd(&M2f[nc * Cc + nr], deg2[nr] * ew[e] * deg2[nc]);
  }
  __syncthreads();
  if (tid < KSEL) M2f[tid * Cc + tid] += deg2[tid] * deg2[tid];
  __syncthreads();
  for (int i = tid; i < Cc * Cc; i += 128) R3[i] = (_Float16)M2f[i];
  __syncthreads();   // M2f retired; R2h may now be overwritten

  // ---- GEMM2: xw2 = xp(64x256) @ W2h^T -> xw2hT[feat][node] ----
  for (int nt = 0; nt < 8; ++nt) {
    v8f acc = {};
#pragma unroll
    for (int kb = 0; kb < 8; ++kb) {
      v16h a = load_frag_A((const _Float16*)R1, NH, 16 * wv, 32 * kb, lane);
      v16h b = load_frag_B(W2h, NH, 16 * nt, 32 * kb, lane);
      acc = wmma_f16(a, b, acc);
    }
#pragma unroll
    for (int v = 0; v < 8; ++v) {
      int node = 16 * wv + v + ((lane >> 4) << 3);
      int f    = 16 * nt + (lane & 15);
      R2h[f * Cc + node] = (_Float16)acc[v];
    }
  }
  __syncthreads();

  // ---- Agg2: x2 = M2 @ xw2, + b2, BN2, PReLU -> global out + node mean ----
  const float alpha2 = a2[0];
  for (int nt = 0; nt < 8; ++nt) {
    v8f acc = {};
#pragma unroll
    for (int kb = 0; kb < 2; ++kb) {
      v16h a = load_frag_A((const _Float16*)R3, Cc, 16 * wv, 32 * kb, lane);
      v16h b = load_frag_B((const _Float16*)R2h, Cc, 16 * nt, 32 * kb, lane);
      acc = wmma_f16(a, b, acc);
    }
    int f = 16 * nt + (lane & 15);
    float sc = g2[f] * rsqrtf(rv2[f] + EPSC);
    float sh = be2[f] - rm2[f] * sc;
    float bv = b2[f];
#pragma unroll
    for (int v = 0; v < 8; ++v) {
      int node = 16 * wv + v + ((lane >> 4) << 3);
      if (node < KSEL) {
        float x = acc[v] + bv;
        x = x * sc + sh;
        x = (x >= 0.f) ? x : alpha2 * x;
        out[(size_t)(g * KSEL + node) * NO + f] = x;
        atomicAdd(&zsum[f], x);
      }
    }
  }
  __syncthreads();

  // ---- z_seq[b, f, t] = mean over nodes ----
  {
    int f = tid;  // 128 threads == NO
    float z = zsum[f] * (1.0f / (float)KSEL);
    out[(size_t)ZOFF + (size_t)bb * (NO * TPn) + f * TPn + tt] = z;
  }
}

// ---------------------------------------------------------------------------
// Launch
// ---------------------------------------------------------------------------
extern "C" void kernel_launch(void* const* d_in, const int* in_sizes, int n_in,
                              void* d_out, int out_size, void* d_ws, size_t ws_size,
                              hipStream_t stream) {
  (void)in_sizes; (void)n_in; (void)out_size; (void)ws_size;
  const float* h   = (const float*)d_in[0];
  const int*   ei  = (const int*)  d_in[1];
  const float* ew  = (const float*)d_in[2];
  const float* W1  = (const float*)d_in[3];
  const float* b1  = (const float*)d_in[4];
  const float* g1  = (const float*)d_in[5];
  const float* be1 = (const float*)d_in[6];
  const float* rm1 = (const float*)d_in[7];
  const float* rv1 = (const float*)d_in[8];
  const float* a1  = (const float*)d_in[9];
  const float* Wp  = (const float*)d_in[10];
  const float* bp  = (const float*)d_in[11];
  const float* W2  = (const float*)d_in[12];
  const float* b2  = (const float*)d_in[13];
  const float* g2  = (const float*)d_in[14];
  const float* be2 = (const float*)d_in[15];
  const float* rm2 = (const float*)d_in[16];
  const float* rv2 = (const float*)d_in[17];
  const float* a2  = (const float*)d_in[18];
  float* out = (float*)d_out;

  char* ws = (char*)d_ws;
  float*    M1f = (float*)ws;                       // 16384 B
  float*    Mpf = (float*)(ws + 16384);             // 16384 B
  _Float16* W1h = (_Float16*)(ws + 32768);          // 114688 B
  _Float16* W2h = (_Float16*)(ws + 32768 + 114688); // 65536 B

  gcn_prep_kernel<<<1, 256, 0, stream>>>(ei, ew, W1, W2, M1f, Mpf, W1h, W2h);
  gcn_main_kernel<<<Gg, 128, SMEM_BYTES, stream>>>(
      h, ei, ew, b1, g1, be1, rm1, rv1, a1, Wp, bp,
      b2, g2, be2, rm2, rv2, a2, M1f, Mpf, W1h, W2h, out);
}
